// DCT_base_Rec_Module_18056042513026
// MI455X (gfx1250) — matrix operand order repaired and checked by reading, not compiled, not run on hardware
//
#include <hip/hip_runtime.h>
#include <hip/hip_bf16.h>

typedef __attribute__((ext_vector_type(16))) _Float16 v16h;
typedef __attribute__((ext_vector_type(8)))  _Float16 v8h;
typedef __attribute__((ext_vector_type(8)))  float    v8f;

#define WS   32
#define NUMP 127          // (2048-32)/16+1
#define NPAT (NUMP*NUMP)  // 16129
#define IMGH 2048

// ---- CDNA5 WMMA 16x16x32 f16 operand layouts (ISA 7.12.2) ----
// A (16x32): lane = half*16 + m ; element e -> K = 8*half + e (e<8),
//            16 + 8*half + (e-8) (e>=8)  => two contiguous 8-half runs.
// B (32x16): lane = half*16 + n ; element e -> K = 8*half + e (e<8),
//            16 + 8*half + (e-8) (e>=8)  => two contiguous 8-half runs.
// C/D (f32): VGPR r -> M = r + 8*half, N = lane&15.

__device__ __forceinline__ v16h concat8(v8h lo, v8h hi) {
    return __builtin_shufflevector(lo, hi, 0, 1, 2, 3, 4, 5, 6, 7,
                                   8, 9, 10, 11, 12, 13, 14, 15);
}

// A[m][k] = src[(tileM+m)*32 + k], src row-major (stride 32, 16B aligned)
__device__ __forceinline__ v16h load_A_rm(const _Float16* src, int tileM) {
    int lane = threadIdx.x & 31;
    int m = lane & 15, half = lane >> 4;
    const _Float16* p = src + (tileM + m) * WS + 8 * half;
    v8h lo = *(const v8h*)(p);
    v8h hi = *(const v8h*)(p + 16);
    return concat8(lo, hi);
}

// B[k][n] = srcCol[(tileN+n)*32 + k], srcCol = column-major image of B
__device__ __forceinline__ v16h load_Bc(const _Float16* srcCol, int tileN) {
    int lane = threadIdx.x & 31;
    int n = lane & 15, half = lane >> 4;
    const _Float16* p = srcCol + (tileN + n) * WS + 8 * half;
    v8h lo = *(const v8h*)(p);
    v8h hi = *(const v8h*)(p + 16);
    return concat8(lo, hi);
}

__device__ __forceinline__ v8f wmma_f16(v16h a, v16h b, v8f c) {
    return __builtin_amdgcn_wmma_f32_16x16x32_f16(
        false, a, false, b, (short)0, c, false, false);
}

// store C tile row-major as f16 (consumed later as A operand)
__device__ __forceinline__ void store_C_rm(_Float16* dst, int ti, int tj, v8f acc) {
    int lane = threadIdx.x & 31;
    int n = lane & 15, half = lane >> 4;
#pragma unroll
    for (int r = 0; r < 8; ++r)
        dst[(ti + r + 8 * half) * WS + tj + n] = (_Float16)acc[r];
}

// fill column-major f16 patch from global f32 rows via float4 loads
__device__ __forceinline__ void fill_patch_cm(_Float16* Pt, const float* xc) {
    int lane = threadIdx.x & 31;
    for (int q = lane; q < 256; q += 32) {
        int row = q >> 3, c4 = (q & 7) << 2;
        const float4 v = *(const float4*)(xc + (size_t)row * IMGH + c4);
        Pt[(c4 + 0) * WS + row] = (_Float16)v.x;
        Pt[(c4 + 1) * WS + row] = (_Float16)v.y;
        Pt[(c4 + 2) * WS + row] = (_Float16)v.z;
        Pt[(c4 + 3) * WS + row] = (_Float16)v.w;
    }
}

// ================= Pass 1: grade every patch =================
__global__ __launch_bounds__(96) void pass1_grade(
    const float* __restrict__ x, const float* __restrict__ dct,
    const float* __restrict__ gmasks, const float* __restrict__ gftnum,
    const float* __restrict__ gweights, float* __restrict__ grade) {
    __shared__ __align__(16) _Float16 sD[WS * WS];   // D row-major  (== col-major of D^T)
    __shared__ __align__(16) _Float16 sDt[WS * WS];  // D^T row-major (== col-major of D)
    __shared__ __align__(16) _Float16 sPt[3][WS * WS];  // patch, column-major
    __shared__ __align__(16) _Float16 sT[3][WS * WS];   // T = D@P, row-major
    __shared__ float sW[WS * WS];
    __shared__ float sRed[3];

    const int tid  = threadIdx.x;
    const int wave = tid >> 5;   // channel 0..2
    const int lane = tid & 31;
    const int l    = blockIdx.x;
    const int pi   = l / NUMP, pj = l % NUMP;

    // D, D^T, and fused band weight w[a,b] = sum_g 2^g*mask_g/ftnum_g
    for (int idx = tid; idx < WS * WS; idx += 96) {
        float d = dct[idx];
        sD[idx] = (_Float16)d;
        sDt[(idx & 31) * WS + (idx >> 5)] = (_Float16)d;
        float w = 0.f;
#pragma unroll
        for (int g = 0; g < 6; ++g)
            w += gmasks[g * WS * WS + idx] * gweights[g] / gftnum[g];
        sW[idx] = w;
    }
    const float* xc = x + (size_t)wave * IMGH * IMGH +
                      (size_t)(pi * 16) * IMGH + (size_t)(pj * 16);
    fill_patch_cm(sPt[wave], xc);
    __syncthreads();

    const int half = lane >> 4, n = lane & 15;
    _Float16* Pt = sPt[wave];
    _Float16* T  = sT[wave];

    // T = D @ P
#pragma unroll
    for (int ti = 0; ti < WS; ti += 16)
#pragma unroll
        for (int tj = 0; tj < WS; tj += 16) {
            v8f acc = {};
            acc = wmma_f16(load_A_rm(sD, ti), load_Bc(Pt, tj), acc);
            store_C_rm(T, ti, tj, acc);
        }

    // X = T @ D^T  (col-major image of D^T is sD) ; fuse grade partial
    float partial = 0.f;
#pragma unroll
    for (int ti = 0; ti < WS; ti += 16)
#pragma unroll
        for (int tj = 0; tj < WS; tj += 16) {
            v8f acc = {};
            acc = wmma_f16(load_A_rm(T, ti), load_Bc(sD, tj), acc);
#pragma unroll
            for (int r = 0; r < 8; ++r) {
                int row = ti + r + 8 * half, col = tj + n;
                partial += __logf(fabsf(acc[r]) + 1.0f) * sW[row * WS + col];
            }
        }

#pragma unroll
    for (int off = 16; off > 0; off >>= 1)
        partial += __shfl_xor(partial, off, 32);
    if (lane == 0) sRed[wave] = partial;
    __syncthreads();
    if (tid == 0) grade[l] = sRed[0] + sRed[1] + sRed[2];
}

// ================= Pass 2: top-2 min / top-2 max =================
__device__ __forceinline__ void ins_min(float& v0, int& i0, float& v1, int& i1,
                                        float v, int i) {
    if (i < 0) return;
    if (v < v0 || (v == v0 && i < i0)) { v1 = v0; i1 = i0; v0 = v; i0 = i; }
    else if (v < v1 || (v == v1 && i < i1)) { v1 = v; i1 = i; }
}
__device__ __forceinline__ void ins_max(float& v0, int& i0, float& v1, int& i1,
                                        float v, int i) {
    if (i < 0) return;
    if (v > v0 || (v == v0 && i > i0)) { v1 = v0; i1 = i0; v0 = v; i0 = i; }
    else if (v > v1 || (v == v1 && i > i1)) { v1 = v; i1 = i; }
}

__global__ __launch_bounds__(256) void pass2_select(
    const float* __restrict__ grade, int n, int* __restrict__ sel) {
    __shared__ float sm0v[256], sm1v[256], sM0v[256], sM1v[256];
    __shared__ int   sm0i[256], sm1i[256], sM0i[256], sM1i[256];
    const int tid = threadIdx.x;
    float m0v = 3.4e38f, m1v = 3.4e38f, M0v = -3.4e38f, M1v = -3.4e38f;
    int   m0i = -1, m1i = -1, M0i = -1, M1i = -1;
    for (int i = tid; i < n; i += 256) {
        float v = grade[i];
        ins_min(m0v, m0i, m1v, m1i, v, i);
        ins_max(M0v, M0i, M1v, M1i, v, i);
    }
    sm0v[tid] = m0v; sm1v[tid] = m1v; sm0i[tid] = m0i; sm1i[tid] = m1i;
    sM0v[tid] = M0v; sM1v[tid] = M1v; sM0i[tid] = M0i; sM1i[tid] = M1i;
    __syncthreads();
    for (int s = 128; s > 0; s >>= 1) {
        if (tid < s) {
            ins_min(m0v, m0i, m1v, m1i, sm0v[tid + s], sm0i[tid + s]);
            ins_min(m0v, m0i, m1v, m1i, sm1v[tid + s], sm1i[tid + s]);
            ins_max(M0v, M0i, M1v, M1i, sM0v[tid + s], sM0i[tid + s]);
            ins_max(M0v, M0i, M1v, M1i, sM1v[tid + s], sM1i[tid + s]);
            sm0v[tid] = m0v; sm1v[tid] = m1v; sm0i[tid] = m0i; sm1i[tid] = m1i;
            sM0v[tid] = M0v; sM1v[tid] = M1v; sM0i[tid] = M0i; sM1i[tid] = M1i;
        }
        __syncthreads();
    }
    if (tid == 0) {
        sel[0] = m0i;  // order[0]
        sel[1] = m1i;  // order[1]
        sel[2] = M0i;  // order[-1]
        sel[3] = M1i;  // order[-2]
    }
}

// ================= Pass 3: band-limited reconstruction for 4 patches =====
__global__ __launch_bounds__(96) void pass3_level(
    const float* __restrict__ x, const float* __restrict__ dct,
    const float* __restrict__ lmask, const int* __restrict__ sel,
    float* __restrict__ out) {
    __shared__ __align__(16) _Float16 sD[WS * WS];
    __shared__ __align__(16) _Float16 sDt[WS * WS];
    __shared__ __align__(16) _Float16 sPt[3][WS * WS];   // patch col-major
    __shared__ __align__(16) _Float16 sXmc[3][WS * WS];  // masked X, col-major
    __shared__ __align__(16) _Float16 sT[3][WS * WS];    // scratch, row-major

    const int tid  = threadIdx.x;
    const int wave = tid >> 5;   // channel
    const int lane = tid & 31;
    const int slot = blockIdx.x; // 0: min0, 1: max0, 2: min1, 3: max1
    const int selIdx = (slot == 0) ? 0 : (slot == 1) ? 2 : (slot == 2) ? 1 : 3;
    const int l  = sel[selIdx];
    const int pi = l / NUMP, pj = l % NUMP;

    for (int idx = tid; idx < WS * WS; idx += 96) {
        float d = dct[idx];
        sD[idx] = (_Float16)d;
        sDt[(idx & 31) * WS + (idx >> 5)] = (_Float16)d;
    }
    const float* xc = x + (size_t)wave * IMGH * IMGH +
                      (size_t)(pi * 16) * IMGH + (size_t)(pj * 16);
    fill_patch_cm(sPt[wave], xc);
    __syncthreads();

    const int half = lane >> 4, n = lane & 15;
    _Float16* Pt  = sPt[wave];
    _Float16* Xmc = sXmc[wave];
    _Float16* T   = sT[wave];

    // T = D @ P  (row-major result: next consumed as A)
#pragma unroll
    for (int ti = 0; ti < WS; ti += 16)
#pragma unroll
        for (int tj = 0; tj < WS; tj += 16) {
            v8f acc = {};
            acc = wmma_f16(load_A_rm(sD, ti), load_Bc(Pt, tj), acc);
            store_C_rm(T, ti, tj, acc);
        }

    // Xm = (T @ D^T) * level_mask, stored column-major: one b128 store/tile
#pragma unroll
    for (int ti = 0; ti < WS; ti += 16)
#pragma unroll
        for (int tj = 0; tj < WS; tj += 16) {
            v8f acc = {};
            acc = wmma_f16(load_A_rm(T, ti), load_Bc(sD, tj), acc);
            v8h pk;
#pragma unroll
            for (int r = 0; r < 8; ++r) {
                int row = ti + r + 8 * half, col = tj + n;
                pk[r] = (_Float16)(acc[r] * lmask[row * WS + col]);
            }
            *(v8h*)(Xmc + (tj + n) * WS + ti + 8 * half) = pk;
        }

    // T2 = D^T @ Xm  (A = D^T row-major = sDt; B col-major = Xmc), reuse sPt
    _Float16* T2 = Pt;
#pragma unroll
    for (int ti = 0; ti < WS; ti += 16)
#pragma unroll
        for (int tj = 0; tj < WS; tj += 16) {
            v8f acc = {};
            acc = wmma_f16(load_A_rm(sDt, ti), load_Bc(Xmc, tj), acc);
            store_C_rm(T2, ti, tj, acc);
        }

    // L = T2 @ D  (B = D, col-major image = sDt) -> f32 output
    float* outc = out + (size_t)slot * 3 * WS * WS + (size_t)wave * WS * WS;
#pragma unroll
    for (int ti = 0; ti < WS; ti += 16)
#pragma unroll
        for (int tj = 0; tj < WS; tj += 16) {
            v8f acc = {};
            acc = wmma_f16(load_A_rm(T2, ti), load_Bc(sDt, tj), acc);
#pragma unroll
            for (int r = 0; r < 8; ++r)
                outc[(ti + r + 8 * half) * WS + tj + n] = acc[r];
        }
}

extern "C" void kernel_launch(void* const* d_in, const int* in_sizes, int n_in,
                              void* d_out, int out_size, void* d_ws, size_t ws_size,
                              hipStream_t stream) {
    const float* x        = (const float*)d_in[0]; // (3,2048,2048)
    const float* dct      = (const float*)d_in[1]; // (32,32)
    const float* lmask    = (const float*)d_in[2]; // (32,32)
    const float* gmasks   = (const float*)d_in[3]; // (6,32,32)
    const float* gftnum   = (const float*)d_in[4]; // (6,)
    const float* gweights = (const float*)d_in[5]; // (6,)
    float* out = (float*)d_out;

    float* grade = (float*)d_ws;                                   // NPAT floats
    int*   sel   = (int*)((char*)d_ws + 16384 * sizeof(float));    // 4 ints

    pass1_grade<<<NPAT, 96, 0, stream>>>(x, dct, gmasks, gftnum, gweights, grade);
    pass2_select<<<1, 256, 0, stream>>>(grade, NPAT, sel);
    pass3_level<<<4, 96, 0, stream>>>(x, dct, lmask, sel, out);
}
